// Transformer_1194000908865
// MI455X (gfx1250) — matrix operand order repaired and checked by reading, not compiled, not run on hardware
//
#include <hip/hip_runtime.h>
#include <hip/hip_bf16.h>

typedef __attribute__((ext_vector_type(16))) _Float16 v16h;
typedef __attribute__((ext_vector_type(8)))  _Float16 v8h;
typedef __attribute__((ext_vector_type(4)))  _Float16 v4h;
typedef __attribute__((ext_vector_type(8)))  float    v8f;

#define Dm 128
#define Hh 8
#define Lq 256
#define Bb 8
#define TOK (Bb*Lq)            // 2048 tokens
#define HD  (Hh*Dm)            // 1024

// ---------------------------------------------------------------------------
// fragment loaders (all-f16 operands, contiguous 16B loads)
// ---------------------------------------------------------------------------
// A 16x32 f16: lane row fixed; e<8 -> K=8*half+e (run at p), e>=8 -> +16 (run at p+16)
__device__ __forceinline__ v16h ldA(const _Float16* p) {
    v8h x0 = *(const v8h*)(p);
    v8h x1 = *(const v8h*)(p + 16);
    return __builtin_shufflevector(x0, x1, 0,1,2,3,4,5,6,7,8,9,10,11,12,13,14,15);
}
// B 32x16 f16 (transposed source): e<8 -> K=8*half+e (run at p), e>=8 -> +16 (run at p+16)
__device__ __forceinline__ v16h ldB(const _Float16* p) {
    v8h x0 = *(const v8h*)(p);
    v8h x1 = *(const v8h*)(p + 16);
    return __builtin_shufflevector(x0, x1, 0,1,2,3,4,5,6,7,8,9,10,11,12,13,14,15);
}

#define WMMA(a, b, c) __builtin_amdgcn_wmma_f32_16x16x32_f16( \
        false, (a), false, (b), (short)0, (c), false, false)

// ---------------------------------------------------------------------------
// Unified all-f16 GEMM: C(64x16 per wave) = A[M,K] * Bt[N,K]^T (+bias)(+relu)
// Double-buffered k-loop, 4 WMMAs per 32-wide k-step.
// (b,h) batch strides for every operand; C stored fp32 or f16, opt transposed.
// ---------------------------------------------------------------------------
template<bool RELU, bool CT, bool C16>
__global__ __launch_bounds__(32)
void gemm_f16(const _Float16* __restrict__ A, const _Float16* __restrict__ Bt,
              const float* __restrict__ bias, void* __restrict__ Cv,
              int K, int lda, int ldb, int ldc, int hcount,
              long long sAb, long long sAh, long long sBb, long long sBh,
              long long sCb, long long sCh, long long sBiasH)
{
    const int z = blockIdx.z;
    const int h = z % hcount;
    const int b = z / hcount;
    A  += (long long)b * sAb + (long long)h * sAh;
    Bt += (long long)b * sBb + (long long)h * sBh;
    if (bias) bias += (long long)h * sBiasH;

    const int n0   = blockIdx.x * 16;
    const int m0   = blockIdx.y * 64;
    const int lane = threadIdx.x;
    const int half = lane >> 4;
    const int l15  = lane & 15;

    v8f c0 = {}, c1 = {}, c2 = {}, c3 = {};
    const _Float16* ar0 = A + (long long)(m0 + l15) * lda + 8 * half;
    const _Float16* ar1 = ar0 + 16LL * lda;
    const _Float16* ar2 = ar0 + 32LL * lda;
    const _Float16* ar3 = ar0 + 48LL * lda;
    const _Float16* br  = Bt + (long long)(n0 + l15) * ldb + 8 * half;

    // prologue: fragments for k = 0
    v16h a0 = ldA(ar0), a1 = ldA(ar1), a2 = ldA(ar2), a3 = ldA(ar3);
    v16h bf = ldB(br);
    for (int k0 = 32; k0 < K; k0 += 32) {
        v16h na0 = ldA(ar0 + k0);
        v16h na1 = ldA(ar1 + k0);
        v16h na2 = ldA(ar2 + k0);
        v16h na3 = ldA(ar3 + k0);
        v16h nbf = ldB(br + k0);
        c0 = WMMA(a0, bf, c0);
        c1 = WMMA(a1, bf, c1);
        c2 = WMMA(a2, bf, c2);
        c3 = WMMA(a3, bf, c3);
        a0 = na0; a1 = na1; a2 = na2; a3 = na3; bf = nbf;
    }
    c0 = WMMA(a0, bf, c0);
    c1 = WMMA(a1, bf, c1);
    c2 = WMMA(a2, bf, c2);
    c3 = WMMA(a3, bf, c3);

    const int n = n0 + l15;
    const float bv = bias ? bias[n] : 0.0f;
    v8f* accs[4] = { &c0, &c1, &c2, &c3 };
    float*    Cf = (float*)Cv    + (long long)b * sCb + (long long)h * sCh;
    _Float16* Ch = (_Float16*)Cv + (long long)b * sCb + (long long)h * sCh;
    #pragma unroll
    for (int t = 0; t < 4; ++t) {
        #pragma unroll
        for (int r = 0; r < 8; ++r) {
            const int m = m0 + t * 16 + r + 8 * half;
            float v = (*accs[t])[r] + bv;
            if (RELU) v = fmaxf(v, 0.0f);
            const long long idx = CT ? ((long long)n * ldc + m)
                                     : ((long long)m * ldc + n);
            if (C16) Ch[idx] = (_Float16)v;
            else     Cf[idx] = v;
        }
    }
}

// ---------------------------------------------------------------------------
// fp32 [K,N] -> f16 transposed [N,K], LDS-tiled (weights, once per launch)
// ---------------------------------------------------------------------------
__global__ __launch_bounds__(256)
void transpose_cvt(const float* __restrict__ in, _Float16* __restrict__ out,
                   int K, int N, long long sIn, long long sOut)
{
    __shared__ _Float16 tile[16][17];
    in  += (long long)blockIdx.z * sIn;
    out += (long long)blockIdx.z * sOut;
    const int tx = threadIdx.x & 15, ty = threadIdx.x >> 4;
    const int k0 = blockIdx.y * 16, n0 = blockIdx.x * 16;
    tile[ty][tx] = (_Float16)in[(long long)(k0 + ty) * N + n0 + tx];
    __syncthreads();
    out[(long long)(n0 + ty) * K + k0 + tx] = tile[tx][ty];
}

// fp32 -> f16 elementwise (activation feed), 4 elements/thread
__global__ __launch_bounds__(256)
void cvt_f16(const float* __restrict__ in, _Float16* __restrict__ out)
{
    const int i = (blockIdx.x * 256 + threadIdx.x) * 4;
    float4 v = *(const float4*)(in + i);
    v4h o; o[0] = (_Float16)v.x; o[1] = (_Float16)v.y;
           o[2] = (_Float16)v.z; o[3] = (_Float16)v.w;
    *(v4h*)(out + i) = o;
}

// ---------------------------------------------------------------------------
// Column softmax over query axis (faithful softmax(axis=2)); mask before scale.
// Writes normalized probs as f16 (AV-GEMM A operand).
// ---------------------------------------------------------------------------
__global__ __launch_bounds__(256)
void col_softmax(float* __restrict__ S, _Float16* __restrict__ P,
                 const unsigned char* __restrict__ mask)
{
    const int z = blockIdx.x;            // b*H + h
    const int b = z / Hh;
    const int j = threadIdx.x;           // key index m (column)
    float* Sp = S + (long long)z * Lq * Lq + j;
    _Float16* Pp = P + (long long)z * Lq * Lq + j;
    const bool msk = mask && (mask[(long long)b * Lq + j] != 0);
    const float invs = 0.08838834764831845f; // 1/sqrt(128)

    float mx = -3.402823466e38f;
    for (int l = 0; l < Lq; ++l) {
        float v = Sp[(long long)l * Lq];
        if (msk) v = -1e9f;
        mx = fmaxf(mx, v);
    }
    float sum = 0.0f;
    for (int l = 0; l < Lq; ++l) {
        float v = Sp[(long long)l * Lq];
        if (msk) v = -1e9f;
        float e = __expf((v - mx) * invs);
        Sp[(long long)l * Lq] = e;
        sum += e;
    }
    const float inv = 1.0f / sum;
    for (int l = 0; l < Lq; ++l)
        Pp[(long long)l * Lq] = (_Float16)(Sp[(long long)l * Lq] * inv);
}

// ---------------------------------------------------------------------------
// LayerNorm over last dim (128) + optional residual; one wave per row.
// ---------------------------------------------------------------------------
__global__ __launch_bounds__(256)
void ln_add(const float* __restrict__ src, const float* __restrict__ res,
            float* __restrict__ dst)
{
    const int wave = threadIdx.x >> 5;
    const int lane = threadIdx.x & 31;
    const int row  = blockIdx.x * 8 + wave;
    float4 v = ((const float4*)(src + (long long)row * Dm))[lane];
    float s = v.x + v.y + v.z + v.w;
    #pragma unroll
    for (int m = 16; m; m >>= 1) s += __shfl_xor(s, m, 32);
    const float mu = s * (1.0f / 128.0f);
    const float dx = v.x - mu, dy = v.y - mu, dz = v.z - mu, dw = v.w - mu;
    float q = dx*dx + dy*dy + dz*dz + dw*dw;
    #pragma unroll
    for (int m = 16; m; m >>= 1) q += __shfl_xor(q, m, 32);
    const float rs = rsqrtf(q * (1.0f / 128.0f) + 1e-6f);
    float4 o; o.x = dx*rs; o.y = dy*rs; o.z = dz*rs; o.w = dw*rs;
    if (res) {
        float4 r = ((const float4*)(res + (long long)row * Dm))[lane];
        o.x += r.x; o.y += r.y; o.z += r.z; o.w += r.w;
    }
    ((float4*)(dst + (long long)row * Dm))[lane] = o;
}

__global__ __launch_bounds__(256)
void add_pe(const float* __restrict__ in, float* __restrict__ out)
{
    const int idx = blockIdx.x * 256 + threadIdx.x;   // over B*L*D
    const int d = idx & (Dm - 1);
    const int l = (idx / Dm) & (Lq - 1);
    const int i2 = d & ~1;
    const float div = __expf(-(float)i2 * (9.210340371976184f / (float)Dm));
    const float ang = (float)l * div;
    const float pe = (d & 1) ? __cosf(ang) : __sinf(ang);
    out[idx] = in[idx] + pe;
}

// ===========================================================================
// Host-side orchestration
// ===========================================================================
namespace {

typedef long long ll;

struct AttB { const float *bq,*bk,*bv,*bo; const float *Wq,*Wk,*Wv,*Wo; };
struct FFB  { const float *b1,*b2,*b3; const float *W1,*W2,*W3; };
struct WS   { float *z,*t1,*t2,*S;
              _Float16 *z16,*Q16,*K16,*V16,*P16,*O16,*t1h,*h2h;
              _Float16 *encA,*encF,*decA1,*decA2,*decF; };

const ll ATT = 524288;  // halves per att set (Wq,Wk,Wv,Wo transposed)
const ll FFS = 32768;   // halves per ff set

AttB att_b(void* const* din, int base, int l) {
    AttB w;
    w.Wq = (const float*)din[base+0] + (ll)l*Hh*Dm*Dm;
    w.bq = (const float*)din[base+1] + (ll)l*Hh*Dm;
    w.Wk = (const float*)din[base+2] + (ll)l*Hh*Dm*Dm;
    w.bk = (const float*)din[base+3] + (ll)l*Hh*Dm;
    w.Wv = (const float*)din[base+4] + (ll)l*Hh*Dm*Dm;
    w.bv = (const float*)din[base+5] + (ll)l*Hh*Dm;
    w.Wo = (const float*)din[base+6] + (ll)l*HD*Dm;
    w.bo = (const float*)din[base+7] + (ll)l*Dm;
    return w;
}
FFB ff_b(void* const* din, int base, int l) {
    FFB w;
    w.W1 = (const float*)din[base+0] + (ll)l*128*128;
    w.b1 = (const float*)din[base+1] + (ll)l*128;
    w.W2 = (const float*)din[base+2] + (ll)l*128*64;
    w.b2 = (const float*)din[base+3] + (ll)l*64;
    w.W3 = (const float*)din[base+4] + (ll)l*64*128;
    w.b3 = (const float*)din[base+5] + (ll)l*128;
    return w;
}

void tconv(hipStream_t s, const float* in, _Float16* out, int K, int N,
           int bat, ll sIn, ll sOut)
{
    dim3 g(N / 16, K / 16, bat);
    transpose_cvt<<<g, 256, 0, s>>>(in, out, K, N, sIn, sOut);
}
void conv_att(hipStream_t s, const AttB& w, _Float16* dst) {
    tconv(s, w.Wq, dst,           128, 128, Hh, 16384, 16384);
    tconv(s, w.Wk, dst + 131072,  128, 128, Hh, 16384, 16384);
    tconv(s, w.Wv, dst + 262144,  128, 128, Hh, 16384, 16384);
    tconv(s, w.Wo, dst + 393216, 1024, 128, 1, 0, 0);
}
void conv_ff(hipStream_t s, const FFB& w, _Float16* dst) {
    tconv(s, w.W1, dst,          128, 128, 1, 0, 0);
    tconv(s, w.W2, dst + 16384,  128,  64, 1, 0, 0);
    tconv(s, w.W3, dst + 24576,   64, 128, 1, 0, 0);
}

void gemm(hipStream_t s, const _Float16* A, const _Float16* Bt, const float* bias,
          void* C, int M, int N, int K, int lda, int ldb, int ldc,
          int bcount, int hcount,
          ll sAb, ll sAh, ll sBb, ll sBh, ll sCb, ll sCh, ll sBiasH,
          bool relu, bool ct, bool c16)
{
    dim3 g(N / 16, M / 64, bcount * hcount);
    if (relu)
        gemm_f16<true, false,true ><<<g,32,0,s>>>(A,Bt,bias,C,K,lda,ldb,ldc,hcount,sAb,sAh,sBb,sBh,sCb,sCh,sBiasH);
    else if (ct)
        gemm_f16<false,true, true ><<<g,32,0,s>>>(A,Bt,bias,C,K,lda,ldb,ldc,hcount,sAb,sAh,sBb,sBh,sCb,sCh,sBiasH);
    else if (c16)
        gemm_f16<false,false,true ><<<g,32,0,s>>>(A,Bt,bias,C,K,lda,ldb,ldc,hcount,sAb,sAh,sBb,sBh,sCb,sCh,sBiasH);
    else
        gemm_f16<false,false,false><<<g,32,0,s>>>(A,Bt,bias,C,K,lda,ldb,ldc,hcount,sAb,sAh,sBb,sBh,sCb,sCh,sBiasH);
}

void mhsa(hipStream_t s, const _Float16* z16, const AttB& w, const _Float16* wt,
          const unsigned char* mask, WS& ws, float* out)
{
    // Q,K,V: per-head projections; Q16/K16 as [B,L,H*128] f16, V transposed [h][e][tok]
    gemm(s, z16, wt,          w.bq, ws.Q16, TOK,128,128, 128,128,HD, 1,Hh,
         0,0, 0,16384, 0,128, 128, false,false,true);
    gemm(s, z16, wt + 131072, w.bk, ws.K16, TOK,128,128, 128,128,HD, 1,Hh,
         0,0, 0,16384, 0,128, 128, false,false,true);
    gemm(s, z16, wt + 262144, w.bv, ws.V16, TOK,128,128, 128,128,TOK, 1,Hh,
         0,0, 0,16384, 0,(ll)128*TOK, 128, false,true,true);
    // scores[b,h,l,m] = Q @ K^T  (fp32 out for softmax)
    gemm(s, ws.Q16, ws.K16, nullptr, ws.S, Lq,Lq,128, HD,HD,Lq, Bb,Hh,
         (ll)Lq*HD,128, (ll)Lq*HD,128, (ll)Hh*Lq*Lq,(ll)Lq*Lq, 0,
         false,false,false);
    col_softmax<<<Bb*Hh, 256, 0, s>>>(ws.S, ws.P16, mask);
    // O = attn @ V  -> f16 [B,L,H*128]
    gemm(s, ws.P16, ws.V16, nullptr, ws.O16, Lq,128,Lq, Lq,TOK,HD, Bb,Hh,
         (ll)Hh*Lq*Lq,(ll)Lq*Lq, 256,(ll)128*TOK, (ll)Lq*HD,128, 0,
         false,false,true);
    // out = O @ Wo + bo  (fp32 out -> LN/residual)
    gemm(s, ws.O16, wt + 393216, w.bo, out, TOK,128,HD, HD,1024,128, 1,1,
         0,0,0,0,0,0,0, false,false,false);
}

void ffn(hipStream_t s, const _Float16* z16, const FFB& w, const _Float16* wt,
         WS& ws, float* out)
{
    gemm(s, z16,    wt,         w.b1, ws.t1h, TOK,128,128, 128,128,128, 1,1,
         0,0,0,0,0,0,0, true, false,true);
    gemm(s, ws.t1h, wt + 16384, w.b2, ws.h2h, TOK, 64,128, 128,128, 64, 1,1,
         0,0,0,0,0,0,0, true, false,true);
    gemm(s, ws.h2h, wt + 24576, w.b3, out,    TOK,128, 64,  64, 64,128, 1,1,
         0,0,0,0,0,0,0, false,false,false);
}

void run_block(hipStream_t s, const float* in, float* out, void* const* din,
               const unsigned char* mask, WS& ws)
{
    const int CVB = TOK * Dm / (256 * 4);
    add_pe<<<TOK * Dm / 256, 256, 0, s>>>(in, ws.z);
    for (int l = 0; l < 7; ++l) {                       // encoder layers
        AttB a = att_b(din, 4, l);
        FFB  f = ff_b(din, 12, l);
        cvt_f16<<<CVB, 256, 0, s>>>(ws.z, ws.z16);
        mhsa(s, ws.z16, a, ws.encA + l*ATT, mask, ws, ws.t1);
        ln_add<<<TOK/8, 256, 0, s>>>(ws.t1, ws.z, ws.z);
        cvt_f16<<<CVB, 256, 0, s>>>(ws.z, ws.z16);
        ffn(s, ws.z16, f, ws.encF + l*FFS, ws, ws.t2);
        ln_add<<<TOK/8, 256, 0, s>>>(ws.t2, ws.z, ws.z);
    }
    ln_add<<<TOK/8, 256, 0, s>>>(ws.z, nullptr, ws.z);  // enc-stack final LN
    for (int l = 0; l < 7; ++l) {                       // decoder layers
        AttB a1 = att_b(din, 18, l);
        AttB a2 = att_b(din, 26, l);
        FFB  f  = ff_b(din, 34, l);
        cvt_f16<<<CVB, 256, 0, s>>>(ws.z, ws.z16);
        mhsa(s, ws.z16, a1, ws.decA1 + l*ATT, mask, ws, ws.t1);
        ln_add<<<TOK/8, 256, 0, s>>>(ws.t1, ws.z, ws.z);
        cvt_f16<<<CVB, 256, 0, s>>>(ws.z, ws.z16);
        mhsa(s, ws.z16, a2, ws.decA2 + l*ATT, mask, ws, ws.t1);
        ln_add<<<TOK/8, 256, 0, s>>>(ws.t1, ws.z, ws.z);
        cvt_f16<<<CVB, 256, 0, s>>>(ws.z, ws.z16);
        ffn(s, ws.z16, f, ws.decF + l*FFS, ws, ws.t2);
        ln_add<<<TOK/8, 256, 0, s>>>(ws.t2, ws.z, ws.z);
    }
    ln_add<<<TOK/8, 256, 0, s>>>(ws.z, nullptr, out);   // dec-stack final LN
}

} // namespace

extern "C" void kernel_launch(void* const* d_in, const int* in_sizes, int n_in,
                              void* d_out, int out_size, void* d_ws, size_t ws_size,
                              hipStream_t stream)
{
    (void)in_sizes; (void)n_in; (void)out_size; (void)ws_size;
    float* w = (float*)d_ws;
    WS ws;
    ll off = 0;
    ws.z  = w + off; off += (ll)TOK * Dm;
    ws.t1 = w + off; off += (ll)TOK * Dm;
    ws.t2 = w + off; off += (ll)TOK * Dm;
    ws.S  = w + off; off += (ll)Bb * Hh * Lq * Lq;

    _Float16* wh = (_Float16*)(w + off);
    ws.z16 = wh;  wh += (ll)TOK * Dm;
    ws.Q16 = wh;  wh += (ll)TOK * HD;
    ws.K16 = wh;  wh += (ll)TOK * HD;
    ws.V16 = wh;  wh += (ll)TOK * HD;
    ws.O16 = wh;  wh += (ll)TOK * HD;
    ws.P16 = wh;  wh += (ll)Bb * Hh * Lq * Lq;
    ws.t1h = wh;  wh += (ll)TOK * Dm;
    ws.h2h = wh;  wh += (ll)TOK * 64;
    ws.encA  = wh; wh += 7 * ATT;
    ws.encF  = wh; wh += 7 * FFS;
    ws.decA1 = wh; wh += 7 * ATT;
    ws.decA2 = wh; wh += 7 * ATT;
    ws.decF  = wh; wh += 7 * FFS;

    // One-time (per launch) weight convert+transpose to f16
    for (int l = 0; l < 7; ++l) {
        conv_att(stream, att_b(d_in, 4,  l), ws.encA  + l*ATT);
        conv_ff (stream, ff_b (d_in, 12, l), ws.encF  + l*FFS);
        conv_att(stream, att_b(d_in, 18, l), ws.decA1 + l*ATT);
        conv_att(stream, att_b(d_in, 26, l), ws.decA2 + l*ATT);
        conv_ff (stream, ff_b (d_in, 34, l), ws.decF  + l*FFS);
    }

    const float* x = (const float*)d_in[0];
    const unsigned char* mask = (const unsigned char*)d_in[1];
    float* out = (float*)d_out;
    float* m0 = out;
    float* m1 = out + (ll)TOK * Dm;
    float* m2 = out + 2LL * TOK * Dm;

    run_block(stream, x,  m0, d_in, mask, ws);
    run_block(stream, m0, m1, d_in, mask, ws);
    run_block(stream, m1, m2, d_in, mask, ws);
}
